// MLP_CNN_41016937677403
// MI455X (gfx1250) — compile-verified, matrix operand order checked by reading, not run on hardware
//
#include <hip/hip_runtime.h>
#include <cstdint>

// ---------- CDNA5 WMMA / TDM types ----------
typedef __attribute__((ext_vector_type(16))) __bf16        v16bf;
typedef __attribute__((ext_vector_type(8)))  __bf16        v8bf;
typedef __attribute__((ext_vector_type(8)))  float         v8f;
typedef __attribute__((ext_vector_type(4)))  unsigned int  v4u;
typedef __attribute__((ext_vector_type(8)))  int           v8i;
typedef __attribute__((ext_vector_type(4)))  int           v4i;

#if defined(__has_builtin)
#if __has_builtin(__builtin_amdgcn_tensor_load_to_lds)
#define USE_TDM 1
#else
#define USE_TDM 0
#endif
#else
#define USE_TDM 0
#endif

// Problem constants
#define N_R     2000
#define N_P     1512
#define M_ROWS  3512
#define MP      3584          // M padded to 128
#define SIZE_R  3000
#define KP_ATT  3008          // 3000 padded to 32
#define FUN_IN  5603
#define KP_FUN  5632          // 5603 padded to 32
#define ATT_H   2048
#define FUN_H   4096
#define E_COLS  512
#define NPAIR   8192
#define PROTEIN 1512

#define ACT_GELU    1
#define ACT_SIGMOID 2

// =====================================================================
// TDM: issue a 128-row x 32-col bf16 tile load, global -> LDS, with
// LDS padding of 4 dwords after every 16 dwords (row stride 40 bf16).
// D# built per CDNA5 ISA (group0 + group1, 2-D).
// This toolchain exposes the 6-arg builtin:
//   (uint32x4 g0, int32x8 g1, int32x4 g2, int32x4 g3, int32x8, i32 cpol)
// =====================================================================
#if USE_TDM
__device__ __forceinline__ void tdm_load_tile(unsigned lds_off,
                                              const __bf16* gptr,
                                              unsigned tensor_d0,   // K extent (elems)
                                              unsigned tensor_d1,   // row count
                                              unsigned long long stride0) // row stride (elems)
{
    const unsigned long long ga = (unsigned long long)gptr;
    v4u g0;
    g0[0] = 1u;                                   // count=1, user descriptor
    g0[1] = lds_off;                              // lds_addr (bytes)
    g0[2] = (unsigned)(ga & 0xFFFFFFFFull);       // global_addr[31:0]
    g0[3] = (unsigned)((ga >> 32) & 0x1FFFFFFull) // global_addr[56:32]
            | (2u << 30);                         // type = 2 ("image")
    v8i g1;
    // data_size=1 (2B), pad_enable, pad_interval=3 (16 dwords = one 32-elem
    // bf16 row), pad_amount=3 (4 dwords = 8 bf16) -> LDS row stride 40 elems
    g1[0] = (int)((1u << 16) | (1u << 20) | (3u << 22) | (3u << 25));
    g1[1] = (int)((tensor_d0 & 0xFFFFu) << 16);                       // dim0 lo
    g1[2] = (int)((tensor_d0 >> 16) | ((tensor_d1 & 0xFFFFu) << 16)); // dim0 hi | dim1 lo
    g1[3] = (int)((tensor_d1 >> 16) | (32u << 16));                   // dim1 hi | tile_dim0=32
    g1[4] = 128;                                                      // tile_dim1=128, tile_dim2=0
    g1[5] = (int)(stride0 & 0xFFFFFFFFull);                           // dim0_stride lo
    g1[6] = (int)((stride0 >> 32) & 0xFFFFull);                       // dim0_stride hi
    g1[7] = 0;
    const v4i z4 = {0, 0, 0, 0};
    const v8i z8 = {0, 0, 0, 0, 0, 0, 0, 0};
    __builtin_amdgcn_tensor_load_to_lds(g0, g1, z4, z4, z8, 0);
}
#endif

// Synchronous fallback staging (also used if TDM builtin is unavailable)
__device__ __forceinline__ void stage_tile_sync(__bf16* dst, const __bf16* src,
                                                size_t stride, int tid)
{
    const int r  = tid >> 1;
    const int kh = (tid & 1) << 4;
    const v8bf* s = (const v8bf*)(src + (size_t)r * stride + kh);
    v8bf*       d = (v8bf*)(dst + r * 40 + kh);
    d[0] = s[0];
    d[1] = s[1];
}

__device__ __forceinline__ v16bf frag_from_lds(const __bf16* rowBase, int g)
{
    const v8bf lo = *(const v8bf*)(rowBase + g * 8);
    const v8bf hi = *(const v8bf*)(rowBase + g * 8 + 16);
    v16bf f;
#pragma unroll
    for (int i = 0; i < 8; ++i) { f[i] = lo[i]; f[i + 8] = hi[i]; }
    return f;
}

// =====================================================================
// Pre-passes: pack concat(A0,A1) f32 -> padded bf16 [Mp x Kp];
//             transpose W (KxN f32) -> Wt (N x Kp bf16), zero-padded.
// Pure bandwidth (~170 MB total): negligible vs GEMM compute.
// =====================================================================
__global__ __launch_bounds__(256)
void concat_row_bf16(const float* __restrict__ A0, const float* __restrict__ A1,
                     int split, int M, int K, int Kp, __bf16* __restrict__ dst)
{
    const int row = blockIdx.x;
    const bool rok = row < M;
    const float* src = (row < split) ? (A0 + (size_t)row * K)
                                     : (A1 + (size_t)(row - split) * K);
    for (int c = threadIdx.x; c < Kp; c += 256)
        dst[(size_t)row * Kp + c] = (__bf16)((rok && c < K) ? src[c] : 0.0f);
}

__global__ __launch_bounds__(256)
void transpose_f32_to_bf16(const float* __restrict__ W, int K, int N, int Kp,
                           __bf16* __restrict__ Wt)
{
    __shared__ float tile[32][33];
    const int kb = blockIdx.y * 32, nb = blockIdx.x * 32;
    const int tx = threadIdx.x & 31, ty = threadIdx.x >> 5;   // 32 x 8
#pragma unroll
    for (int r = ty; r < 32; r += 8) {
        const int k = kb + r;
        tile[r][tx] = (k < K) ? W[(size_t)k * N + nb + tx] : 0.0f;
    }
    __syncthreads();
#pragma unroll
    for (int r = ty; r < 32; r += 8)
        Wt[(size_t)(nb + r) * Kp + kb + tx] = (__bf16)tile[tx][r];
}

// =====================================================================
// GEMM:  C = act(A @ Wt^T + bias)   A: Mp x Kp bf16, Wt: N x Kp bf16.
// Block 128x128, 8 wave32 waves, wave tile 32x64 (2x4 WMMA frags).
// Double-buffered LDS fed by TDM (tensor_load_to_lds + s_wait_tensorcnt).
// OUT_BF16=1: write bf16 hidden activations (pad rows zeroed).
// OUT_BF16=0: f32 output with concat-split routing at column colOff.
// =====================================================================
template<int ACT, int OUT_BF16>
__global__ __launch_bounds__(256)
void gemm_tdm_wmma(const __bf16* __restrict__ A, const __bf16* __restrict__ Bt,
                   int Mp, int Kp, int N, int M,
                   const float* __restrict__ bias,
                   float* __restrict__ C0, float* __restrict__ C1,
                   int outSplit, int ldc0, int ldc1, int colOff,
                   __bf16* __restrict__ Hout)
{
    __shared__ __bf16 sA[2][128 * 40];
    __shared__ __bf16 sB[2][128 * 40];

    const int tid  = threadIdx.x;
    const int wave = tid >> 5, lane = tid & 31;
    const int g    = lane >> 4, lr = lane & 15;
    const int wm   = wave & 3,  wn = wave >> 2;
    const int mBlock = blockIdx.y * 128;
    const int nBlock = blockIdx.x * 128;

    const __bf16* Arow = A  + (size_t)mBlock * Kp;
    const __bf16* Brow = Bt + (size_t)nBlock * Kp;

    v8f acc[2][4] = {};
    const int kTiles = Kp >> 5;

#if USE_TDM
    const unsigned aOff[2] = { (unsigned)(unsigned long long)&sA[0][0],
                               (unsigned)(unsigned long long)&sA[1][0] };
    const unsigned bOff[2] = { (unsigned)(unsigned long long)&sB[0][0],
                               (unsigned)(unsigned long long)&sB[1][0] };
    if (wave == 0) {
        tdm_load_tile(aOff[0], Arow, (unsigned)Kp, (unsigned)Mp, (unsigned long long)Kp);
        tdm_load_tile(bOff[0], Brow, (unsigned)Kp, (unsigned)N,  (unsigned long long)Kp);
    }
#endif

    for (int kt = 0; kt < kTiles; ++kt) {
        const int cur = kt & 1;
#if USE_TDM
        if (wave == 0) __builtin_amdgcn_s_wait_tensorcnt(0);
        __syncthreads();                       // publish TDM data to all waves
        if (wave == 0 && kt + 1 < kTiles) {    // prefetch next tile via DMA
            const int nk = (kt + 1) << 5;
            tdm_load_tile(aOff[cur ^ 1], Arow + nk, (unsigned)Kp, (unsigned)Mp,
                          (unsigned long long)Kp);
            tdm_load_tile(bOff[cur ^ 1], Brow + nk, (unsigned)Kp, (unsigned)N,
                          (unsigned long long)Kp);
        }
#else
        stage_tile_sync(sA[cur], Arow + (kt << 5), (size_t)Kp, tid);
        stage_tile_sync(sB[cur], Brow + (kt << 5), (size_t)Kp, tid);
        __syncthreads();
#endif
        v16bf a[2], b[4];
#pragma unroll
        for (int mi = 0; mi < 2; ++mi)
            a[mi] = frag_from_lds(&sA[cur][(wm * 32 + mi * 16 + lr) * 40], g);
#pragma unroll
        for (int ni = 0; ni < 4; ++ni)
            b[ni] = frag_from_lds(&sB[cur][(wn * 64 + ni * 16 + lr) * 40], g);

#pragma unroll
        for (int mi = 0; mi < 2; ++mi)
#pragma unroll
            for (int ni = 0; ni < 4; ++ni)
                acc[mi][ni] = __builtin_amdgcn_wmma_f32_16x16x32_bf16(
                    false, a[mi], false, b[ni], (short)0, acc[mi][ni],
                    false, false);
#if !USE_TDM
        __syncthreads();
#endif
    }

    // ---- epilogue ----
#pragma unroll
    for (int ni = 0; ni < 4; ++ni) {
        const int col = nBlock + wn * 64 + ni * 16 + lr;
        const float bv = bias[col];
#pragma unroll
        for (int mi = 0; mi < 2; ++mi) {
#pragma unroll
            for (int v = 0; v < 8; ++v) {
                const int row = mBlock + wm * 32 + mi * 16 + g * 8 + v;
                float x = acc[mi][ni][v] + bv;
                if (ACT == ACT_GELU)
                    x = 0.5f * x * (1.0f + erff(x * 0.70710678118654752f));
                else
                    x = 1.0f / (1.0f + __expf(-x));
                if (OUT_BF16) {
                    Hout[(size_t)row * N + col] = (__bf16)((row < M) ? x : 0.0f);
                } else if (row < M) {
                    float* Cp = (row < outSplit)
                                  ? (C0 + (size_t)row * ldc0)
                                  : (C1 + (size_t)(row - outSplit) * ldc1);
                    Cp[colOff + col] = x;
                }
            }
        }
    }
}

// =====================================================================
// Per-pair CNN: conv1 VALU, conv2 as implicit-im2col WMMA GEMM
// (M=1024, K=256 padded, N=32), maxpool+tanh, nontemporal flat store,
// fused 8192x2 matvec reduction.
// =====================================================================
#define LDS_XROW  0
#define LDS_RED   4096
#define LDS_W1    6144
#define LDS_B1    7168
#define LDS_B2    7232
#define LDS_H1    7360
#define LDS_WB    (7360 + 16384)
#define LDS_OUT2  (7360 + 32768)
#define CNN_LDS_BYTES (LDS_OUT2 + 65536)

__global__ __launch_bounds__(256)
void cnn_pair_wmma(const int* __restrict__ idx,
                   const float* __restrict__ e_r, const float* __restrict__ e_p,
                   const float* __restrict__ c1w, const float* __restrict__ c1b,
                   const float* __restrict__ c2w, const float* __restrict__ c2b,
                   const float* __restrict__ Wout, const float* __restrict__ bout,
                   float* __restrict__ flat, float* __restrict__ outp)
{
    extern __shared__ char smem[];
    float*  xrow = (float*)(smem + LDS_XROW);
    float*  red  = (float*)(smem + LDS_RED);
    float*  w1   = (float*)(smem + LDS_W1);
    float*  b1s  = (float*)(smem + LDS_B1);
    float*  b2s  = (float*)(smem + LDS_B2);
    __bf16* h1   = (__bf16*)(smem + LDS_H1);
    __bf16* wB   = (__bf16*)(smem + LDS_WB);
    __bf16* out2 = (__bf16*)(smem + LDS_OUT2);

    const int tid = threadIdx.x;
    const int pb  = blockIdx.x;
    const int id   = idx[pb];
    const int r_no = id / PROTEIN;
    const int p_no = id % PROTEIN;

    for (int i = tid; i < 512; i += 256) {
        xrow[i]       = e_r[(size_t)r_no * E_COLS + i];
        xrow[512 + i] = e_p[(size_t)p_no * E_COLS + i];
    }
    for (int i = tid; i < 240; i += 256) w1[i] = c1w[i];
    if (tid < 16) b1s[tid] = c1b[tid];
    if (tid < 32) b2s[tid] = c2b[tid];
    for (int i = tid; i < 32 * 256; i += 256) {
        const int n = i >> 8, k = i & 255;
        const int ci = k >> 4, tap = k & 15;
        const float v = (tap < 15) ? c2w[(n * 16 + ci) * 15 + tap] : 0.0f;
        wB[n * 256 + k] = (__bf16)v;
    }
    __syncthreads();

    for (int o = tid; o < 16 * 2 * 256; o += 256) {
        const int co = o >> 9, rem = o & 511;
        const int hp = rem >> 8, wp = rem & 255;
        float s = 0.0f;
#pragma unroll
        for (int dh = 0; dh < 2; ++dh) {
            const int ho = hp * 2 + dh;
#pragma unroll
            for (int dw = 0; dw < 2; ++dw) {
                const int wo = wp * 2 + dw;
                float acc = b1s[co];
#pragma unroll
                for (int kh = 0; kh < 3; ++kh) {
                    const int hi = ho + kh - 2;
                    if (hi < 0 || hi >= 2) continue;
#pragma unroll
                    for (int kw = 0; kw < 5; ++kw) {
                        const int wi = wo + kw - 2;
                        if (wi < 0 || wi >= 512) continue;
                        acc += w1[co * 15 + kh * 5 + kw] * xrow[hi * 512 + wi];
                    }
                }
                s += (acc >= 0.0f) ? acc : 0.01f * acc;
            }
        }
        h1[o] = (__bf16)(s * 0.25f);
    }
    __syncthreads();

    const int wave = tid >> 5, lane = tid & 31;
    const int g = lane >> 4, lr = lane & 15;
    for (int j = 0; j < 8; ++j) {
        const int mt = wave * 8 + j;
        const int m  = mt * 16 + lr;
        const int ho = m >> 8, wo = m & 255;
        v8f acc0 = {}, acc1 = {};
#pragma unroll
        for (int ks = 0; ks < 8; ++ks) {
            v16bf a;
#pragma unroll
            for (int i = 0; i < 16; ++i) {
                const int k   = ks * 32 + g * 8 + i + ((i >= 8) ? 8 : 0);
                const int ci  = k >> 4, tap = k & 15;
                const int kh  = tap / 5, kw = tap - kh * 5;
                const int hi  = ho + kh - 2, wi = wo + kw - 2;
                const bool ok = (tap < 15) && (hi >= 0) && (hi < 2) &&
                                (wi >= 0) && (wi < 256);
                a[i] = ok ? h1[ci * 512 + hi * 256 + wi] : (__bf16)0.0f;
            }
            v16bf bA = frag_from_lds(&wB[lr * 256 + ks * 32], g);
            v16bf bB = frag_from_lds(&wB[(16 + lr) * 256 + ks * 32], g);
            acc0 = __builtin_amdgcn_wmma_f32_16x16x32_bf16(
                false, a, false, bA, (short)0, acc0, false, false);
            acc1 = __builtin_amdgcn_wmma_f32_16x16x32_bf16(
                false, a, false, bB, (short)0, acc1, false, false);
        }
#pragma unroll
        for (int v = 0; v < 8; ++v) {
            const int mm = mt * 16 + g * 8 + v;
            const float x0 = acc0[v] + b2s[lr];
            const float x1 = acc1[v] + b2s[16 + lr];
            out2[mm * 32 + lr]      = (__bf16)((x0 >= 0.0f) ? x0 : 0.01f * x0);
            out2[mm * 32 + 16 + lr] = (__bf16)((x1 >= 0.0f) ? x1 : 0.01f * x1);
        }
    }
    __syncthreads();

    float s0 = 0.0f, s1 = 0.0f;
    for (int o = tid; o < NPAIR; o += 256) {
        const int co = o >> 8, rem = o & 255;
        const int hp = rem >> 7, wp = rem & 127;
        const int m00 = hp * 512 + wp * 2;
        const float v0 = (float)out2[(m00      ) * 32 + co];
        const float v1 = (float)out2[(m00 +   1) * 32 + co];
        const float v2 = (float)out2[(m00 + 256) * 32 + co];
        const float v3 = (float)out2[(m00 + 257) * 32 + co];
        const float t  = tanhf(fmaxf(fmaxf(v0, v1), fmaxf(v2, v3)));
        __builtin_nontemporal_store(t, &flat[(size_t)pb * NPAIR + o]);
        s0 += t * Wout[o * 2];
        s1 += t * Wout[o * 2 + 1];
    }
    red[tid] = s0; red[256 + tid] = s1;
    __syncthreads();
    for (int s = 128; s > 0; s >>= 1) {
        if (tid < s) { red[tid] += red[tid + s]; red[256 + tid] += red[256 + tid + s]; }
        __syncthreads();
    }
    if (tid == 0) {
        outp[(size_t)pb * 2 + 0] = red[0]   + bout[0];
        outp[(size_t)pb * 2 + 1] = red[256] + bout[1];
    }
}

// =====================================================================
// Launcher
// =====================================================================
extern "C" void kernel_launch(void* const* d_in, const int* in_sizes, int n_in,
                              void* d_out, int out_size, void* d_ws, size_t ws_size,
                              hipStream_t stream)
{
    const float* r_att  = (const float*)d_in[0];
    const float* p_att  = (const float*)d_in[1];
    const float* r_fun  = (const float*)d_in[2];
    const float* p_fun  = (const float*)d_in[3];
    const int*   idx    = (const int*)  d_in[4];
    const float* W_att1 = (const float*)d_in[5];
    const float* b_att1 = (const float*)d_in[6];
    const float* W_att2 = (const float*)d_in[7];
    const float* b_att2 = (const float*)d_in[8];
    const float* W_fun1 = (const float*)d_in[9];
    const float* b_fun1 = (const float*)d_in[10];
    const float* W_fun2 = (const float*)d_in[11];
    const float* b_fun2 = (const float*)d_in[12];
    const float* c1w    = (const float*)d_in[13];
    const float* c1b    = (const float*)d_in[14];
    const float* c2w    = (const float*)d_in[15];
    const float* c2b    = (const float*)d_in[16];
    const float* W_out  = (const float*)d_in[17];
    const float* b_out  = (const float*)d_in[18];

    float* out  = (float*)d_out;
    float* e_r  = out;
    float* e_p  = e_r + (size_t)N_R * E_COLS;
    float* outp = e_p + (size_t)N_P * E_COLS;
    float* flat = outp + (size_t)NPAIR * 2;

    // workspace layout (bf16)
    __bf16* w = (__bf16*)d_ws;
    __bf16* Abf_att = w;                       w += (size_t)MP * KP_ATT;
    __bf16* Abf_fun = w;                       w += (size_t)MP * KP_FUN;
    __bf16* Wt_att1 = w;                       w += (size_t)ATT_H * KP_ATT;
    __bf16* Wt_att2 = w;                       w += (size_t)256 * ATT_H;
    __bf16* Wt_fun1 = w;                       w += (size_t)FUN_H * KP_FUN;
    __bf16* Wt_fun2 = w;                       w += (size_t)256 * FUN_H;
    __bf16* H_att   = w;                       w += (size_t)MP * ATT_H;
    __bf16* H_fun   = w;                       w += (size_t)MP * FUN_H;

    const dim3 blk(256);
    const int mGrid = MP / 128;   // 28

    // pack / convert / transpose pre-passes (bandwidth-trivial)
    concat_row_bf16<<<MP, blk, 0, stream>>>(r_att, p_att, N_R, M_ROWS, SIZE_R, KP_ATT, Abf_att);
    concat_row_bf16<<<MP, blk, 0, stream>>>(r_fun, p_fun, N_R, M_ROWS, FUN_IN, KP_FUN, Abf_fun);
    transpose_f32_to_bf16<<<dim3(ATT_H / 32, KP_ATT / 32), blk, 0, stream>>>(W_att1, SIZE_R, ATT_H, KP_ATT, Wt_att1);
    transpose_f32_to_bf16<<<dim3(256   / 32, ATT_H  / 32), blk, 0, stream>>>(W_att2, ATT_H, 256, ATT_H, Wt_att2);
    transpose_f32_to_bf16<<<dim3(FUN_H / 32, KP_FUN / 32), blk, 0, stream>>>(W_fun1, FUN_IN, FUN_H, KP_FUN, Wt_fun1);
    transpose_f32_to_bf16<<<dim3(256   / 32, FUN_H  / 32), blk, 0, stream>>>(W_fun2, FUN_H, 256, FUN_H, Wt_fun2);

    // att branch
    gemm_tdm_wmma<ACT_GELU, 1><<<dim3(ATT_H / 128, mGrid), blk, 0, stream>>>(
        Abf_att, Wt_att1, MP, KP_ATT, ATT_H, M_ROWS, b_att1,
        nullptr, nullptr, 0, 0, 0, 0, H_att);
    gemm_tdm_wmma<ACT_SIGMOID, 0><<<dim3(256 / 128, mGrid), blk, 0, stream>>>(
        H_att, Wt_att2, MP, ATT_H, 256, M_ROWS, b_att2,
        e_r, e_p, N_R, E_COLS, E_COLS, 0, nullptr);

    // fun branch
    gemm_tdm_wmma<ACT_GELU, 1><<<dim3(FUN_H / 128, mGrid), blk, 0, stream>>>(
        Abf_fun, Wt_fun1, MP, KP_FUN, FUN_H, M_ROWS, b_fun1,
        nullptr, nullptr, 0, 0, 0, 0, H_fun);
    gemm_tdm_wmma<ACT_SIGMOID, 0><<<dim3(256 / 128, mGrid), blk, 0, stream>>>(
        H_fun, Wt_fun2, MP, FUN_H, 256, M_ROWS, b_fun2,
        e_r, e_p, N_R, E_COLS, E_COLS, 256, nullptr);

    // pair CNN
    cnn_pair_wmma<<<dim3(NPAIR), blk, CNN_LDS_BYTES, stream>>>(
        idx, e_r, e_p, c1w, c1b, c2w, c2b, W_out, b_out, flat, outp);
}
